// SafeRoutingLinear_9869834846652
// MI455X (gfx1250) — matrix operand order repaired and checked by reading, not compiled, not run on hardware
//
#include <hip/hip_runtime.h>
#include <hip/hip_bf16.h>

#define NTOK 2048
#define KDIM 1024
#define NOUT 50257
#define TOPK 32
#define NEG_INF (-3.402823466e38f)

typedef __attribute__((ext_vector_type(16))) __bf16 v16bf;
typedef __attribute__((ext_vector_type(8)))  float  v8f;

union FragU { v16bf v; uint4 u[2]; };

// ---------------------------------------------------------------------------
// Kernel 1: inverse L2 norm of each weight row (matches F.normalize eps=1e-12)
// ---------------------------------------------------------------------------
__global__ __launch_bounds__(256) void wnorm_kernel(const float* __restrict__ w,
                                                    float* __restrict__ invnw) {
    __shared__ float red[256];
    const int o = blockIdx.x;
    const int t = threadIdx.x;
    const float4* row = (const float4*)(w + (size_t)o * KDIM);
    float4 v = row[t];                       // 256 * 4 = 1024 elements
    red[t] = v.x * v.x + v.y * v.y + v.z * v.z + v.w * v.w;
    __syncthreads();
    for (int s = 128; s > 0; s >>= 1) {
        if (t < s) red[t] += red[t + s];
        __syncthreads();
    }
    if (t == 0) {
        float n = sqrtf(red[0]);
        invnw[o] = 1.0f / fmaxf(n, 1e-12f);
    }
}

// ---------------------------------------------------------------------------
// Kernel 2: dense D = x @ w^T, bf16 WMMA with f32 accumulation.
// BM=128, BN=128, BK=32; 256 threads = 8 waves, each wave owns 32x64.
// Double-buffered LDS: next K-step's global loads are issued before the
// current step's WMMAs so the load latency overlaps the matrix pipes.
// D is written into d_out ([2048, 50257] f32).
// ---------------------------------------------------------------------------
__device__ __forceinline__ void stage_load(const float* __restrict__ x,
                                           const float* __restrict__ w,
                                           int m0, int n0, int k0, int tid,
                                           float4 aR[4], float4 bR[4]) {
#pragma unroll
    for (int c = 0; c < 4; ++c) {
        int chunk = tid + 256 * c;             // 0..1023
        int r     = chunk >> 3;                // 0..127
        int kc    = (chunk & 7) * 4;           // 0..28
        aR[c] = *(const float4*)(x + (size_t)(m0 + r) * KDIM + k0 + kc);
        int nrow = n0 + r;
        int crow = nrow < NOUT ? nrow : (NOUT - 1);   // clamped: always legal
        bR[c] = *(const float4*)(w + (size_t)crow * KDIM + k0 + kc);
    }
}

__device__ __forceinline__ void stage_store(__bf16* __restrict__ bufA,
                                            __bf16* __restrict__ bufB,
                                            int tid, const float4 aR[4],
                                            const float4 bR[4],
                                            const float mask[4]) {
#pragma unroll
    for (int c = 0; c < 4; ++c) {
        int chunk = tid + 256 * c;
        int r     = chunk >> 3;
        int kc    = (chunk & 7) * 4;
        __bf16* pa = &bufA[r * 32 + kc];
        pa[0] = (__bf16)aR[c].x; pa[1] = (__bf16)aR[c].y;
        pa[2] = (__bf16)aR[c].z; pa[3] = (__bf16)aR[c].w;
        float m = mask[c];                     // branchless OOB zeroing
        __bf16* pb = &bufB[r * 32 + kc];
        pb[0] = (__bf16)(bR[c].x * m); pb[1] = (__bf16)(bR[c].y * m);
        pb[2] = (__bf16)(bR[c].z * m); pb[3] = (__bf16)(bR[c].w * m);
    }
}

__global__ __launch_bounds__(256) void gemm_kernel(const float* __restrict__ x,
                                                   const float* __restrict__ w,
                                                   float* __restrict__ out) {
    __shared__ __align__(16) __bf16 ldsA[2][128 * 32];   // 2 x 8 KB
    __shared__ __align__(16) __bf16 ldsB[2][128 * 32];   // 2 x 8 KB

    const int tid  = threadIdx.x;
    const int lane = tid & 31;
    const int wave = tid >> 5;
    const int wm   = (wave & 3) * 32;
    const int wn   = (wave >> 2) * 64;
    const int m0   = blockIdx.y * 128;
    const int n0   = blockIdx.x * 128;
    const int half = lane >> 4;
    const int l16  = lane & 15;

    // Per-chunk OOB mask for the B tile (rows fixed per thread for all K).
    float mask[4];
#pragma unroll
    for (int c = 0; c < 4; ++c) {
        int r = (tid + 256 * c) >> 3;
        mask[c] = (n0 + r < NOUT) ? 1.0f : 0.0f;
    }

    v8f acc[2][4];
#pragma unroll
    for (int i = 0; i < 2; ++i)
#pragma unroll
        for (int j = 0; j < 4; ++j)
            acc[i][j] = (v8f){0.f, 0.f, 0.f, 0.f, 0.f, 0.f, 0.f, 0.f};

    float4 aR[4], bR[4];
    stage_load(x, w, m0, n0, 0, tid, aR, bR);
    stage_store(ldsA[0], ldsB[0], tid, aR, bR, mask);
    __syncthreads();

    int p = 0;
    for (int k0 = 0; k0 < KDIM; k0 += 32) {
        const bool hasNext = (k0 + 32) < KDIM;
        if (hasNext)                       // issue next tile's loads early;
            stage_load(x, w, m0, n0, k0 + 32, tid, aR, bR);  // wait lands post-WMMA

        // --- compute on buffer p ---
        const __bf16* bufA = ldsA[p];
        const __bf16* bufB = ldsB[p];
        FragU a[2];
#pragma unroll
        for (int i = 0; i < 2; ++i) {
            int row  = wm + i * 16 + l16;
            int koff = half * 8;
            a[i].u[0] = *(const uint4*)&bufA[row * 32 + koff];
            a[i].u[1] = *(const uint4*)&bufA[row * 32 + koff + 16];
        }
        FragU b[4];
#pragma unroll
        for (int j = 0; j < 4; ++j) {
            int col = wn + j * 16 + l16;
            int kb  = half * 8;
            b[j].u[0] = *(const uint4*)&bufB[col * 32 + kb];
            b[j].u[1] = *(const uint4*)&bufB[col * 32 + kb + 16];
        }
#pragma unroll
        for (int i = 0; i < 2; ++i)
#pragma unroll
            for (int j = 0; j < 4; ++j)
                acc[i][j] = __builtin_amdgcn_wmma_f32_16x16x32_bf16(
                    false, a[i].v, false, b[j].v,
                    (short)0, acc[i][j], false, false);

        if (hasNext) {
            __syncthreads();               // everyone done reading buf p^1
            stage_store(ldsA[p ^ 1], ldsB[p ^ 1], tid, aR, bR, mask);
            __syncthreads();               // new tile visible to all
            p ^= 1;
        }
    }

    // C/D layout: VGPR r holds M = r + 8*half, N = l16
#pragma unroll
    for (int i = 0; i < 2; ++i) {
        int rbase = m0 + wm + i * 16 + half * 8;
#pragma unroll
        for (int j = 0; j < 4; ++j) {
            int col = n0 + wn + j * 16 + l16;
            if (col < NOUT) {
#pragma unroll
                for (int r = 0; r < 8; ++r)
                    out[(size_t)(rbase + r) * NOUT + col] = acc[i][j][r];
            }
        }
    }
}

// ---------------------------------------------------------------------------
// Kernel 3: per-token top-32 by (D * invnw) then in-place sparse rewrite.
// One block of 128 threads per token row. invnx (positive per-row scale)
// cannot change the row's top-k ordering, so it is skipped entirely.
// ---------------------------------------------------------------------------
__global__ __launch_bounds__(128) void topk_kernel(float* __restrict__ out,
                                                   const float* __restrict__ invnw,
                                                   const float* __restrict__ bias) {
    __shared__ float cvals[128 * 32];    // 16 KB candidate scores
    __shared__ int   cids [128 * 32];    // 16 KB candidate ids
    __shared__ float rv[128];
    __shared__ int   ri[128];
    __shared__ int   topid[TOPK];
    __shared__ float topval[TOPK];

    const int n = blockIdx.x;
    const int t = threadIdx.x;
    float* row = out + (size_t)n * NOUT;

    // Phase 1: per-thread private top-32 over strided chunk (~393 elems).
    float lv[TOPK];
    int   li[TOPK];
    for (int i = 0; i < TOPK; ++i) { lv[i] = NEG_INF; li[i] = 0; }
    float lmin = NEG_INF;
    int   lmi  = 0;
    for (int o = t; o < NOUT; o += 128) {
        float s = row[o] * invnw[o];
        if (s > lmin) {
            lv[lmi] = s; li[lmi] = o;
            lmin = lv[0]; lmi = 0;
            for (int i = 1; i < TOPK; ++i)
                if (lv[i] < lmin) { lmin = lv[i]; lmi = i; }
        }
    }
    for (int i = 0; i < TOPK; ++i) {
        cvals[t * TOPK + i] = lv[i];
        cids [t * TOPK + i] = li[i];
    }
    __syncthreads();

    // Phase 2: 32 rounds of block-wide argmax over the 4096 candidates.
    for (int k = 0; k < TOPK; ++k) {
        float mv = NEG_INF;
        int   mi = t * TOPK;
        for (int i = 0; i < TOPK; ++i) {
            float v = cvals[t * TOPK + i];
            if (v > mv) { mv = v; mi = t * TOPK + i; }
        }
        rv[t] = mv; ri[t] = mi;
        __syncthreads();
        for (int s = 64; s > 0; s >>= 1) {
            if (t < s && rv[t + s] > rv[t]) { rv[t] = rv[t + s]; ri[t] = ri[t + s]; }
            __syncthreads();
        }
        if (t == 0) {
            int e = ri[0];
            topid[k] = cids[e];
            cvals[e] = NEG_INF;      // consume winner
        }
        __syncthreads();
    }

    // Phase 3: capture raw D + bias at winners, zero the row, scatter.
    if (t < TOPK) {
        int id = topid[t];
        topval[t] = row[id] + bias[id];
    }
    __syncthreads();
    for (int o = t; o < NOUT; o += 128) row[o] = 0.0f;
    __threadfence_block();
    __syncthreads();
    if (t < TOPK) row[topid[t]] = topval[t];
}

// ---------------------------------------------------------------------------
extern "C" void kernel_launch(void* const* d_in, const int* in_sizes, int n_in,
                              void* d_out, int out_size, void* d_ws, size_t ws_size,
                              hipStream_t stream) {
    const float* x    = (const float*)d_in[0];   // [2,1024,1024]
    const float* w    = (const float*)d_in[1];   // [50257,1024]
    const float* bias = (const float*)d_in[2];   // [50257]
    float* out   = (float*)d_out;                // [2048, 50257]
    float* invnw = (float*)d_ws;                 // 50257 floats scratch

    wnorm_kernel<<<NOUT, 256, 0, stream>>>(w, invnw);

    dim3 grid((NOUT + 127) / 128, NTOK / 128);   // (393, 16)
    gemm_kernel<<<grid, 256, 0, stream>>>(x, w, out);

    topk_kernel<<<NTOK, 128, 0, stream>>>(out, invnw, bias);
}
